// GRU4RecModel_44487271252271
// MI455X (gfx1250) — compile-verified, matrix operand order checked by reading, not run on hardware
//
#include <hip/hip_runtime.h>
#include <hip/hip_bf16.h>

// ---------------------------------------------------------------------------
// GRU4Rec on MI455X (gfx1250): bf16 WMMA (f32 accumulate) for all GEMMs,
// persistent per-batch-slice scan kernel for the recurrent part.
// B=1024, T=200, H=256, 3H=768, 2 layers. K is always 256.
// ---------------------------------------------------------------------------

#define Hdim    256
#define G3      768
#define T_STEPS 200
#define B_SZ    1024
#define KDIM    256

typedef __attribute__((ext_vector_type(16))) __bf16 v16bf;
typedef __attribute__((ext_vector_type(8)))  float  v8f;

union Frag16 { v16bf v; uint4 q[2]; };

__device__ __forceinline__ v8f v8f_zero() {
  v8f z = {0.f, 0.f, 0.f, 0.f, 0.f, 0.f, 0.f, 0.f};
  return z;
}

// round-to-nearest-even f32 -> bf16 bits
__device__ __forceinline__ unsigned int f2bf_bits(float f) {
  union { float f; unsigned int u; } c; c.f = f;
  unsigned int u = c.u;
  u += 0x7FFFu + ((u >> 16) & 1u);
  return u >> 16;
}

// Load one 16x32 bf16 WMMA operand fragment from a row-major [*,256] array.
// ISA layout: lane L holds row (L&15); K values klo..klo+7 (VGPR0-3) and
// klo+16..klo+23 (VGPR4-7), klo = kc + (L>=16 ? 8 : 0). Two b128 loads.
__device__ __forceinline__ v16bf load_frag(const unsigned short* row, int klo) {
  Frag16 f;
  f.q[0] = *(const uint4*)(row + klo);
  f.q[1] = *(const uint4*)(row + klo + 16);
  return f.v;
}

// --------------------------- weight convert --------------------------------
__global__ void __launch_bounds__(256) k_f32_to_bf16(const float* __restrict__ in,
                                                     unsigned short* __restrict__ out,
                                                     int n) {
  int i = blockIdx.x * 256 + threadIdx.x;
  if (i < n) out[i] = (unsigned short)f2bf_bits(in[i]);
}

// ------------------------ embedding gather + bf16 --------------------------
// one block per (b,t) row; 64 threads x 4 floats = 256 cols
__global__ void __launch_bounds__(64) k_embed_gather(const int* __restrict__ ids,
                                                     const float* __restrict__ emb,
                                                     unsigned short* __restrict__ x0) {
  size_t bt = blockIdx.x;
  int id = ids[bt];
  const float4* src = (const float4*)(emb + (size_t)id * Hdim);
  float4 v = src[threadIdx.x];
  uint2 o;
  o.x = f2bf_bits(v.x) | (f2bf_bits(v.y) << 16);
  o.y = f2bf_bits(v.z) | (f2bf_bits(v.w) << 16);
  ((uint2*)(x0 + bt * Hdim))[threadIdx.x] = o;
}

// ------------------------------ WMMA GEMM ----------------------------------
// C[M,N] (f32) = A[M,K=256](bf16 row-major) * W[N,K=256]^T (bf16) + bias
// Block = 256 threads = 8 waves, covering 256 M rows. Each wave owns 2 M-tiles
// (32 rows): all 8 K-chunk A fragments for both tiles are pre-loaded into
// registers (A read from L2 exactly once), then the wave sweeps every N-tile
// with a double-buffered B fragment so global_load_b128 of chunk kc+1 overlaps
// the two WMMAs of chunk kc.
template <bool HAS_BIAS>
__global__ void __launch_bounds__(256) k_wmma_gemm(const unsigned short* __restrict__ A,
                                                   const unsigned short* __restrict__ W,
                                                   float* __restrict__ C,
                                                   const float* __restrict__ bias,
                                                   int N) {
  const int lane = threadIdx.x & 31;
  const int wave = threadIdx.x >> 5;
  const int r  = lane & 15;
  const int hi = lane >> 4;
  const int m0 = blockIdx.x * 256 + wave * 32;

  Frag16 a[2][8];
  {
    const unsigned short* arow0 = A + (size_t)(m0 + r) * KDIM;
    const unsigned short* arow1 = A + (size_t)(m0 + 16 + r) * KDIM;
#pragma unroll
    for (int kc = 0; kc < 8; ++kc) {
      const int klo = kc * 32 + hi * 8;
      a[0][kc].q[0] = *(const uint4*)(arow0 + klo);
      a[0][kc].q[1] = *(const uint4*)(arow0 + klo + 16);
      a[1][kc].q[0] = *(const uint4*)(arow1 + klo);
      a[1][kc].q[1] = *(const uint4*)(arow1 + klo + 16);
    }
  }

  const int ntiles = N >> 4;
  for (int nt = 0; nt < ntiles; ++nt) {
    const int n0 = nt << 4;
    const unsigned short* wrow = W + (size_t)(n0 + r) * KDIM;

    Frag16 b[2];                                   // double-buffered B fragment
    b[0].q[0] = *(const uint4*)(wrow + hi * 8);
    b[0].q[1] = *(const uint4*)(wrow + hi * 8 + 16);

    v8f acc0 = v8f_zero();
    v8f acc1 = v8f_zero();
#pragma unroll
    for (int kc = 0; kc < 8; ++kc) {
      if (kc < 7) {                                // prefetch next chunk's B
        const int klo = (kc + 1) * 32 + hi * 8;
        b[(kc + 1) & 1].q[0] = *(const uint4*)(wrow + klo);
        b[(kc + 1) & 1].q[1] = *(const uint4*)(wrow + klo + 16);
      }
      acc0 = __builtin_amdgcn_wmma_f32_16x16x32_bf16(false, a[0][kc].v, false, b[kc & 1].v,
                                                     (short)0, acc0, false, false);
      acc1 = __builtin_amdgcn_wmma_f32_16x16x32_bf16(false, a[1][kc].v, false, b[kc & 1].v,
                                                     (short)0, acc1, false, false);
    }

    float bv = 0.0f;
    if (HAS_BIAS) bv = bias[n0 + r];
    // D layout: n = lane&15 ; m = (lane>=16 ? 8 : 0) + i
    float* c0 = C + (size_t)(m0 + hi * 8) * N + (n0 + r);
    float* c1 = C + (size_t)(m0 + 16 + hi * 8) * N + (n0 + r);
#pragma unroll
    for (int i = 0; i < 8; ++i) {
      c0[(size_t)i * N] = acc0[i] + bv;
      c1[(size_t)i * N] = acc1[i] + bv;
    }
  }
}

// ------------------------------ GRU scan -----------------------------------
// 64 blocks x 256 threads. Block owns batch rows [b0, b0+16).
// Per step: h_proj = h_bf16 @ w_hh^T (48 N-tiles; wave w -> cols [w*96,w*96+96)),
// stash in LDS, then fused gate math; h kept fp32 in regs + bf16 mirror in LDS.
__global__ void __launch_bounds__(256) k_gru_scan(const float* __restrict__ xproj,
                                                  const unsigned short* __restrict__ whh,
                                                  unsigned short* __restrict__ y) {
  __shared__ __align__(16) unsigned short hbf[16][264];   // bf16 h mirror (padded)
  __shared__ float hproj[16][776];                        // fp32 h @ whh^T (padded)

  const int tid  = threadIdx.x;
  const int lane = tid & 31;
  const int wave = tid >> 5;
  const int b0   = blockIdx.x << 4;

  // gate-phase ownership: 16 consecutive cols of one batch row per thread
  const int bl = tid >> 4;            // 0..15 local batch row
  const int j0 = (tid & 15) << 4;     // 0,16,...,240

  float hreg[16];
#pragma unroll
  for (int i = 0; i < 16; ++i) { hreg[i] = 0.0f; hbf[bl][j0 + i] = 0; }
  __syncthreads();

  const int r  = lane & 15;
  const int hi = lane >> 4;
  const int n0base = wave * 96;
  const size_t xrow0 = (size_t)(b0 + bl) * T_STEPS * G3;

  for (int t = 0; t < T_STEPS; ++t) {
    // ---- recurrent GEMM: 6 tiles x 8 K-chunks of v_wmma per wave ----
    v8f acc[6];
#pragma unroll
    for (int s = 0; s < 6; ++s) acc[s] = v8f_zero();
#pragma unroll
    for (int kc = 0; kc < Hdim; kc += 32) {
      const int klo = kc + hi * 8;
      Frag16 af;
      af.q[0] = *(const uint4*)&hbf[r][klo];
      af.q[1] = *(const uint4*)&hbf[r][klo + 16];
#pragma unroll
      for (int s = 0; s < 6; ++s) {
        const unsigned short* wrow = whh + (size_t)(n0base + s * 16 + r) * Hdim;
        v16bf b = load_frag(wrow, klo);
        acc[s] = __builtin_amdgcn_wmma_f32_16x16x32_bf16(false, af.v, false, b,
                                                         (short)0, acc[s], false, false);
      }
    }
#pragma unroll
    for (int s = 0; s < 6; ++s) {
      const int col = n0base + s * 16 + r;
#pragma unroll
      for (int i = 0; i < 8; ++i) hproj[hi * 8 + i][col] = acc[s][i];
    }
    __syncthreads();

    // ---- fused gates: r,z,n -> h_new ----
    const float* xp = xproj + xrow0 + (size_t)t * G3;
    __builtin_prefetch(xp + G3, 0, 0);      // global_prefetch_b8: next timestep
    unsigned int packed[8];
#pragma unroll
    for (int i = 0; i < 16; ++i) {
      const int j = j0 + i;
      const float xr = xp[j], xz = xp[Hdim + j], xn = xp[2 * Hdim + j];
      const float hr = hproj[bl][j], hz = hproj[bl][Hdim + j], hn = hproj[bl][2 * Hdim + j];
      const float rg = 1.0f / (1.0f + __expf(-(xr + hr)));
      const float zg = 1.0f / (1.0f + __expf(-(xz + hz)));
      const float ng = tanhf(xn + rg * hn);
      const float hv = (1.0f - zg) * ng + zg * hreg[i];
      hreg[i] = hv;
      const unsigned int bb = f2bf_bits(hv);
      hbf[bl][j] = (unsigned short)bb;
      if (i & 1) packed[i >> 1] |= bb << 16; else packed[i >> 1] = bb;
    }
    unsigned short* yrow = y + ((size_t)(b0 + bl) * T_STEPS + t) * Hdim + j0;
    *(uint4*)(yrow)     = *(const uint4*)&packed[0];
    *(uint4*)(yrow + 8) = *(const uint4*)&packed[4];
    __syncthreads();
  }
}

// ------------------------------- launcher ----------------------------------
extern "C" void kernel_launch(void* const* d_in, const int* in_sizes, int n_in,
                              void* d_out, int out_size, void* d_ws, size_t ws_size,
                              hipStream_t stream) {
  const int*   ids     = (const int*)d_in[0];
  const float* emb     = (const float*)d_in[1];
  const float* w_ih_l0 = (const float*)d_in[2];
  const float* w_hh_l0 = (const float*)d_in[3];
  const float* w_ih_l1 = (const float*)d_in[4];
  const float* w_hh_l1 = (const float*)d_in[5];
  const float* dense_w = (const float*)d_in[6];
  const float* dense_b = (const float*)d_in[7];
  float* out = (float*)d_out;

  const size_t BT = (size_t)B_SZ * T_STEPS;           // 204800
  char* ws = (char*)d_ws;

  // workspace layout (256B aligned)
  const size_t SZ_W   = (size_t)G3 * Hdim * 2;        // 384 KB (bf16 GRU weight)
  const size_t SZ_DW  = (size_t)Hdim * Hdim * 2;      // 128 KB
  const size_t SZ_ACT = BT * Hdim * 2;                // 100 MB (bf16 activations)
  const size_t SZ_XP  = BT * G3 * 4;                  // 600 MB (f32 x_proj)
  size_t off = 0;
  unsigned short* w_ih0_bf = (unsigned short*)(ws + off); off += SZ_W;
  unsigned short* w_hh0_bf = (unsigned short*)(ws + off); off += SZ_W;
  unsigned short* w_ih1_bf = (unsigned short*)(ws + off); off += SZ_W;
  unsigned short* w_hh1_bf = (unsigned short*)(ws + off); off += SZ_W;
  unsigned short* dw_bf    = (unsigned short*)(ws + off); off += SZ_DW + 256;
  off = (off + 255) & ~(size_t)255;
  unsigned short* x0_bf = (unsigned short*)(ws + off); off += SZ_ACT;
  float*          xproj = (float*)(ws + off);          off += SZ_XP;
  unsigned short* y0_bf = (unsigned short*)(ws + off); off += SZ_ACT;
  unsigned short* y1_bf = (unsigned short*)(ws + off); off += SZ_ACT;
  (void)ws_size; (void)in_sizes; (void)n_in; (void)out_size;

  // 1) weights -> bf16
  const int nW = G3 * Hdim, nD = Hdim * Hdim;
  k_f32_to_bf16<<<(nW + 255) / 256, 256, 0, stream>>>(w_ih_l0, w_ih0_bf, nW);
  k_f32_to_bf16<<<(nW + 255) / 256, 256, 0, stream>>>(w_hh_l0, w_hh0_bf, nW);
  k_f32_to_bf16<<<(nW + 255) / 256, 256, 0, stream>>>(w_ih_l1, w_ih1_bf, nW);
  k_f32_to_bf16<<<(nW + 255) / 256, 256, 0, stream>>>(w_hh_l1, w_hh1_bf, nW);
  k_f32_to_bf16<<<(nD + 255) / 256, 256, 0, stream>>>(dense_w, dw_bf, nD);

  // 2) embedding gather -> bf16 x0
  k_embed_gather<<<(unsigned)BT, 64, 0, stream>>>(ids, emb, x0_bf);

  const int gemm_blocks = (int)(BT / 256);            // 800

  // 3) layer 0: x_proj = x0 @ w_ih0^T   [204800 x 768]
  k_wmma_gemm<false><<<gemm_blocks, 256, 0, stream>>>(x0_bf, w_ih0_bf, xproj, nullptr, G3);
  // 4) layer 0 recurrent scan -> y0
  k_gru_scan<<<B_SZ / 16, 256, 0, stream>>>(xproj, w_hh0_bf, y0_bf);

  // 5) layer 1: x_proj = y0 @ w_ih1^T
  k_wmma_gemm<false><<<gemm_blocks, 256, 0, stream>>>(y0_bf, w_ih1_bf, xproj, nullptr, G3);
  // 6) layer 1 recurrent scan -> y1
  k_gru_scan<<<B_SZ / 16, 256, 0, stream>>>(xproj, w_hh1_bf, y1_bf);

  // 7) dense: out = y1 @ dense_w^T + b  [204800 x 256], fp32 out
  k_wmma_gemm<true><<<gemm_blocks, 256, 0, stream>>>(y1_bf, dw_bf, out, dense_b, Hdim);
}
